// FreqSincExtractor_35897336660489
// MI455X (gfx1250) — matrix operand order repaired and checked by reading, not compiled, not run on hardware
//
#include <hip/hip_runtime.h>
#include <hip/hip_bf16.h>
#include <math.h>

typedef __attribute__((ext_vector_type(16))) _Float16 v16h;
typedef __attribute__((ext_vector_type(8)))  float    v8f;

#define NBATCH   1024
#define NSAMP    16000
#define NFREQ    201        // rfft bins
#define NFRM     101        // STFT frames
#define NFRM_P   112        // 7*16 (GEMM2 N)
#define MT1P     14         // 14*16 = 224 freq rows (M padded, even # of tiles)
#define KS1      13         // GEMM1 K steps of 32 (K padded 400->416)
#define NT2      7          // GEMM2 N tiles (112 frames)
#define MT2      4          // 64 channels
#define KS2      7          // GEMM2 K steps of 32 (K = 224 = MT1P*16, exact)
#define MAGW     224
#define XP_LEN   20736      // = 127*160 + 415 + 1 ; padded frame buffer (f16)
#define HOPL     160
#define NFFT     400

// dynamic LDS layout (bytes)
#define OFF_XP    0                        // 20736 halfs = 41472 B; later reused as filt (64*112 f32 = 28672 B)
#define OFF_MAGT  41472                    // magT: 128*224 halfs = 57344 B (time-major, K-contiguous)
#define OFF_RED   (41472 + 57344)          // 512 floats = 2048 B
#define SMEM_BYTES (41472 + 57344 + 2048)  // 100864 B

#define WMMA16(A,B,C) __builtin_amdgcn_wmma_f32_16x16x32_f16(false, (A), false, (B), (short)0, (C), false, false)
#define LA(p, ks) (*(const v16h*)((p) + (size_t)(ks) * 512))
#define LB(p, ks) (*(const v16h*)((p) + (ks) * 32))

// ---------------------------------------------------------------------------
// Windowed DFT basis, pre-swizzled into WMMA A-fragment lane order.
// Element (mt, ks, lane, j):  f = mt*16 + (lane&15),  k = ks*32 + j + 16*(lane>>4)
// ---------------------------------------------------------------------------
__global__ void dft_tables_kernel(_Float16* __restrict__ crSw, _Float16* __restrict__ ciSw) {
    int i = blockIdx.x * blockDim.x + threadIdx.x;
    const int total = MT1P * KS1 * 32 * 16;    // 93184
    if (i >= total) return;
    int j    = i & 15;
    int lane = (i >> 4) & 31;
    int ks   = (i >> 9) % KS1;
    int mt   = (i >> 9) / KS1;
    int f = mt * 16 + (lane & 15);
    int k = ks * 32 + j + 16 * (lane >> 4);
    float cr = 0.f, ci = 0.f;
    if (f < NFREQ && k < NFFT) {
        const float twopi = 6.28318530717958647692f;
        float w  = 0.5f - 0.5f * __cosf(twopi * (float)k / 400.f);   // periodic Hann
        int   ph = (f * k) % 400;                                     // exact phase reduction
        float a  = (twopi / 400.f) * (float)ph;
        cr =  __cosf(a) * w;
        ci = -__sinf(a) * w;
    }
    crSw[i] = (_Float16)cr;
    ciSw[i] = (_Float16)ci;
}

// ---------------------------------------------------------------------------
// Triangular filterbank: plain f32 copy (second reference output) + f16
// WMMA A-fragment swizzle for GEMM2.
// ---------------------------------------------------------------------------
__device__ __forceinline__ float filt_val(const float* low_hz, const float* band_hz, int c, int f) {
    float low  = 50.f + fabsf(low_hz[c]);
    float high = low + 50.f + fabsf(band_hz[c]);
    high = fminf(fmaxf(high, 50.f), 8000.f);
    float freq   = 40.f * (float)f;            // linspace(0, 8000, 201)
    float center = 0.5f * (low + high);
    float bw     = high - low;
    float resp   = 1.f - fabsf(freq - center) / (0.5f * bw);
    return (freq >= low && freq <= high) ? resp : 0.f;
}

__global__ void filters_kernel(const float* __restrict__ low_hz, const float* __restrict__ band_hz,
                               float* __restrict__ filters_out, _Float16* __restrict__ filtSw) {
    for (int i = threadIdx.x; i < 64 * NFREQ; i += blockDim.x) {
        int c = i / NFREQ, f = i % NFREQ;
        filters_out[i] = filt_val(low_hz, band_hz, c, f);
    }
    const int total = MT2 * KS2 * 32 * 16;     // 14336
    for (int i = threadIdx.x; i < total; i += blockDim.x) {
        int j    = i & 15;
        int lane = (i >> 4) & 31;
        int ks   = (i >> 9) % KS2;
        int mt   = (i >> 9) / KS2;
        int c = mt * 16 + (lane & 15);
        int f = ks * 32 + j + 16 * (lane >> 4);
        filtSw[i] = (_Float16)((f < NFREQ) ? filt_val(low_hz, band_hz, c, f) : 0.f);
    }
}

// ---------------------------------------------------------------------------
// Main fused kernel: one workgroup (8 wave32) per batch element.
// ---------------------------------------------------------------------------
__launch_bounds__(256)
__global__ void sinc_main_kernel(const float* __restrict__ wav,
                                 const float* __restrict__ lnw,
                                 const float* __restrict__ lnb,
                                 const _Float16* __restrict__ crSw,
                                 const _Float16* __restrict__ ciSw,
                                 const _Float16* __restrict__ filtSw,
                                 float* __restrict__ out) {
    extern __shared__ char smem_raw[];
    _Float16* xp   = (_Float16*)(smem_raw + OFF_XP);     // padded waveform, f16
    _Float16* magT = (_Float16*)(smem_raw + OFF_MAGT);   // magT[t][fbin], K-contiguous for GEMM2 B
    float*    filt = (float*)(smem_raw + OFF_XP);        // reuses xp region after GEMM1
    float*    red  = (float*)(smem_raw + OFF_RED);

    const int b    = blockIdx.x;
    const int tid  = threadIdx.x;
    const int lane = tid & 31;
    const int wave = tid >> 5;
    const int l    = lane & 15;
    const int half = lane >> 4;

    // ---- Stage 0: reflect-pad waveform -> LDS f16 (zero tail) ----
    const float* x = wav + (size_t)b * NSAMP;
    for (int i = tid; i < XP_LEN; i += 256) {
        float v = 0.f;
        if (i < NSAMP + 400) {               // valid reflect-padded region [0, 16400)
            int p = i - 200;
            if (p < 0) p = -p;
            if (p > NSAMP - 1) p = 2 * (NSAMP - 1) - p;
            v = x[p];
        }
        xp[i] = (_Float16)v;
    }
    __syncthreads();

    // ---- Stage 1: DFT as GEMM, 2x2 register-blocked, manual double buffering ----
    // macro-tile grid: 7 (M pairs) x 4 (N pairs) = 28 units, covering 224 x 128 output.
    // Accumulators stay in fixed registers (WMMA accumulates in place); the two
    // fragment buffer sets (c*/e*) are refilled directly by memory loads, so no
    // accumulator rotation movs and no WMMA->VALU hazard NOPs in the steady state.
    for (int unit = wave; unit < 28; unit += 8) {
        int mp = unit >> 2, np = unit & 3;
        int mt0 = mp * 2, nt0 = np * 2;
        v8f aR00 = {}, aI00 = {}, aR01 = {}, aI01 = {};
        v8f aR10 = {}, aI10 = {}, aR11 = {}, aI11 = {};
        const _Float16* crp0 = crSw + ((size_t)(mt0 * KS1) * 32 + lane) * 16;
        const _Float16* cip0 = ciSw + ((size_t)(mt0 * KS1) * 32 + lane) * 16;
        const _Float16* crp1 = crp0 + (size_t)KS1 * 512;   // mt0+1
        const _Float16* cip1 = cip0 + (size_t)KS1 * 512;
        const int t0 = nt0 * 16 + l;
        const _Float16* bp0 = xp + t0 * HOPL + 16 * half;  // frame t0
        const _Float16* bp1 = bp0 + 16 * HOPL;             // frame t0+16

        v16h c0r = LA(crp0, 0), c0i = LA(cip0, 0), c1r = LA(crp1, 0), c1i = LA(cip1, 0);
        v16h cb0 = LB(bp0, 0),  cb1 = LB(bp1, 0);
        #pragma unroll 1
        for (int ks = 0; ks < KS1 - 1; ks += 2) {
            v16h e0r = LA(crp0, ks + 1), e0i = LA(cip0, ks + 1);
            v16h e1r = LA(crp1, ks + 1), e1i = LA(cip1, ks + 1);
            v16h eb0 = LB(bp0, ks + 1),  eb1 = LB(bp1, ks + 1);
            aR00 = WMMA16(c0r, cb0, aR00); aI00 = WMMA16(c0i, cb0, aI00);
            aR01 = WMMA16(c0r, cb1, aR01); aI01 = WMMA16(c0i, cb1, aI01);
            aR10 = WMMA16(c1r, cb0, aR10); aI10 = WMMA16(c1i, cb0, aI10);
            aR11 = WMMA16(c1r, cb1, aR11); aI11 = WMMA16(c1i, cb1, aI11);
            c0r = LA(crp0, ks + 2); c0i = LA(cip0, ks + 2);
            c1r = LA(crp1, ks + 2); c1i = LA(cip1, ks + 2);
            cb0 = LB(bp0, ks + 2);  cb1 = LB(bp1, ks + 2);
            aR00 = WMMA16(e0r, eb0, aR00); aI00 = WMMA16(e0i, eb0, aI00);
            aR01 = WMMA16(e0r, eb1, aR01); aI01 = WMMA16(e0i, eb1, aI01);
            aR10 = WMMA16(e1r, eb0, aR10); aI10 = WMMA16(e1i, eb0, aI10);
            aR11 = WMMA16(e1r, eb1, aR11); aI11 = WMMA16(e1i, eb1, aI11);
        }
        // tail: ks = 12 lives in the c* buffers
        aR00 = WMMA16(c0r, cb0, aR00); aI00 = WMMA16(c0i, cb0, aI00);
        aR01 = WMMA16(c0r, cb1, aR01); aI01 = WMMA16(c0i, cb1, aI01);
        aR10 = WMMA16(c1r, cb0, aR10); aI10 = WMMA16(c1i, cb0, aI10);
        aR11 = WMMA16(c1r, cb1, aR11); aI11 = WMMA16(c1i, cb1, aI11);

        const int n0 = nt0 * 16 + l;
        const int n1 = n0 + 16;
        #pragma unroll
        for (int r = 0; r < 8; ++r) {
            int m0 = mt0 * 16 + r + 8 * half;             // f32 C layout: M = r + 8*laneHalf
            int m1 = m0 + 16;
            float re, im;
            re = aR00[r]; im = aI00[r]; magT[n0 * MAGW + m0] = (_Float16)sqrtf(re * re + im * im);
            re = aR10[r]; im = aI10[r]; magT[n0 * MAGW + m1] = (_Float16)sqrtf(re * re + im * im);
            re = aR01[r]; im = aI01[r]; magT[n1 * MAGW + m0] = (_Float16)sqrtf(re * re + im * im);
            re = aR11[r]; im = aI11[r]; magT[n1 * MAGW + m1] = (_Float16)sqrtf(re * re + im * im);
        }
    }
    __syncthreads();

    // ---- Stage 2: filterbank GEMM (64 x 112 x 224) ----
    for (int tile = wave; tile < MT2 * NT2; tile += 8) {
        int mt = tile / NT2, nt = tile % NT2;
        v8f acc = {};
        const _Float16* fp = filtSw + ((size_t)(mt * KS2) * 32 + lane) * 16;
        const int n = nt * 16 + l;
        const _Float16* bp = magT + n * MAGW + 16 * half;
        for (int ks = 0; ks < KS2; ++ks) {
            v16h aF = LA(fp, ks);
            v16h bM = LB(bp, ks);
            acc = WMMA16(aF, bM, acc);
        }
        #pragma unroll
        for (int r = 0; r < 8; ++r) {
            int c = mt * 16 + r + 8 * half;
            filt[c * NFRM_P + n] = acc[r];               // overlays dead xp region
        }
    }
    __syncthreads();

    // ---- Stage 3: global layer-norm stats over (64 ch x 101 frames) ----
    float s = 0.f, s2 = 0.f;
    for (int i = tid; i < 64 * NFRM_P; i += 256) {
        if ((i % NFRM_P) < NFRM) { float v = filt[i]; s += v; s2 += v * v; }
    }
    red[tid] = s; red[256 + tid] = s2;
    __syncthreads();
    for (int off = 128; off > 0; off >>= 1) {
        if (tid < off) {
            red[tid]       += red[tid + off];
            red[256 + tid] += red[256 + tid + off];
        }
        __syncthreads();
    }
    const float inv_n = 1.f / (64.f * 101.f);
    const float mean  = red[0] * inv_n;
    const float var   = red[256] * inv_n - mean * mean;
    const float rstd  = rsqrtf(var + 1e-5f);

    // ---- Stage 4: norm -> square -> log10 -> linear resample 101->64 -> transpose ----
    float* outB = out + (size_t)b * 4096;
    for (int i = tid; i < 4096; i += 256) {
        int tout = i >> 6, c = i & 63;                   // out[b,0,tout,c] = logE[c, interp(tout)]
        float wgt = lnw[c], bia = lnb[c];
        float pos = ((float)tout + 0.5f) * (101.f / 64.f) - 0.5f;
        pos = fminf(fmaxf(pos, 0.f), 100.f);
        int i0 = (int)pos;
        int i1 = i0 + 1; if (i1 > 100) i1 = 100;
        float fr = pos - (float)i0;
        float y0 = wgt * (filt[c * NFRM_P + i0] - mean) * rstd + bia;
        float y1 = wgt * (filt[c * NFRM_P + i1] - mean) * rstd + bia;
        float le0 = __log10f(y0 * y0 + 1e-6f);
        float le1 = __log10f(y1 * y1 + 1e-6f);
        outB[i] = le0 * (1.f - fr) + fr * le1;
    }
}

// ---------------------------------------------------------------------------
extern "C" void kernel_launch(void* const* d_in, const int* in_sizes, int n_in,
                              void* d_out, int out_size, void* d_ws, size_t ws_size,
                              hipStream_t stream) {
    const float* wav  = (const float*)d_in[0];
    const float* low  = (const float*)d_in[1];
    const float* band = (const float*)d_in[2];
    const float* lnw  = (const float*)d_in[3];
    const float* lnb  = (const float*)d_in[4];
    float* out = (float*)d_out;

    // workspace: CrSw (93184 h) | CiSw (93184 h) | FiltSw (14336 h)  = ~401 KB
    _Float16* crSw   = (_Float16*)d_ws;
    _Float16* ciSw   = crSw + (size_t)MT1P * KS1 * 512;
    _Float16* filtSw = ciSw + (size_t)MT1P * KS1 * 512;

    float* filters_out = out + (size_t)NBATCH * 4096;    // second tuple output

    const int tblTotal = MT1P * KS1 * 512;               // 93184
    dft_tables_kernel<<<(tblTotal + 255) / 256, 256, 0, stream>>>(crSw, ciSw);
    filters_kernel<<<1, 256, 0, stream>>>(low, band, filters_out, filtSw);
    sinc_main_kernel<<<NBATCH, 256, SMEM_BYTES, stream>>>(wav, lnw, lnb, crSw, ciSw, filtSw, out);
}
